// StackSAModulePyramid_24068996726846
// MI455X (gfx1250) — matrix operand order repaired and checked by reading, not compiled, not run on hardware
//
#include <hip/hip_runtime.h>
#include <cstddef>

// ---------------------------------------------------------------------------
// StackSAModulePyramid for MI455X (gfx1250, wave32, WMMA)
//   Per level:
//     1. ball_gather       : ball query + packed bf16 gather -> G (rows x 64)
//     2. gemm_bf16_64      : H = G @ Wa^T  (v_wmma_f32_16x16x32_bf16)
//     3. col_stats (2-stage coalesced deterministic reduction)
//     4. bn_relu_bf16      : G <- bf16(relu(BN(H)))  (vectorized b128)
//     5. gemm_bf16_64      : H = G @ Wb^T
//     6. col_stats
//     7. bn_relu_maxpool_out -> strided slice of d_out
// ---------------------------------------------------------------------------

typedef __attribute__((ext_vector_type(16))) __bf16 v16bf;
typedef __attribute__((ext_vector_type(8)))  float  v8f;

union FragBF { v16bf v; uint4 q[2]; };

#define EPSV   1e-5f
#define C_IN   32
#define NPTS   8192
#define BATCH  2
#define TOTCOLS 91        // 27 + 64 concat along axis 1
#define STAT_BLOCKS 256   // stage-1 reduction blocks

// pack two floats into two bf16 in one dword (RNE via __bf16 cast)
__device__ inline unsigned pk2(float a, float b) {
    union { __bf16 h[2]; unsigned u; } x;
    x.h[0] = (__bf16)a; x.h[1] = (__bf16)b;
    return x.u;
}

// Build one packed 64-channel bf16 row: [rel_xyz | 32 feats | zeros]
__device__ inline void build_row(const float* __restrict__ feat, int pj,
                                 float rx, float ry, float rz, uint4 q[8])
{
    float f[32];
    const float4* fp = (const float4*)(feat + (size_t)pj * C_IN);
    #pragma unroll
    for (int i = 0; i < 8; ++i) {
        const float4 v = fp[i];
        f[4*i+0] = v.x; f[4*i+1] = v.y; f[4*i+2] = v.z; f[4*i+3] = v.w;
    }
    unsigned w[32];
    w[0] = pk2(rx, ry);
    w[1] = pk2(rz, f[0]);
    #pragma unroll
    for (int i = 2; i < 17; ++i) w[i] = pk2(f[2*i-3], f[2*i-2]);
    w[17] = pk2(f[31], 0.f);
    #pragma unroll
    for (int i = 18; i < 32; ++i) w[i] = 0u;
    #pragma unroll
    for (int i = 0; i < 8; ++i)
        q[i] = make_uint4(w[4*i], w[4*i+1], w[4*i+2], w[4*i+3]);
}

// ---------------------------------------------------------------------------
// Ball query + gather. One thread per query; points staged via LDS tiles in
// ascending index order ("first ns in-radius" semantics). Accepted rows are
// packed and stored immediately (8x b128) -- no scratch index array; the
// first accepted row is kept in registers for slot back-fill.
// ---------------------------------------------------------------------------
template<int NS>
__global__ void ball_gather_kernel(const float* __restrict__ xyz,
                                   const float* __restrict__ feat,
                                   const float* __restrict__ q,
                                   const float* __restrict__ qr,
                                   __bf16* __restrict__ G,
                                   int Mq)
{
    constexpr int TILE = 1024;
    __shared__ float sx[TILE], sy[TILE], sz[TILE];

    const int qid   = blockIdx.x * blockDim.x + threadIdx.x;   // global query id
    const int batch = (blockIdx.x * blockDim.x) / Mq;          // Mq % blockDim == 0
    const int pbase = batch * NPTS;

    const float qx = q[qid * 3 + 0];
    const float qy = q[qid * 3 + 1];
    const float qz = q[qid * 3 + 2];
    const float r  = qr[qid];
    const float r2 = r * r;

    uint4* grow = (uint4*)(G + (size_t)qid * NS * 64);   // 8 uint4 per row
    uint4 row0q[8];
    int cnt = 0;

    for (int t0 = 0; t0 < NPTS; t0 += TILE) {
        for (int i = threadIdx.x; i < TILE; i += blockDim.x) {
            const int p = (pbase + t0 + i) * 3;
            sx[i] = xyz[p + 0];
            sy[i] = xyz[p + 1];
            sz[i] = xyz[p + 2];
        }
        __syncthreads();
        if (cnt < NS) {
            for (int i = 0; i < TILE; ++i) {
                const float dx = sx[i] - qx;
                const float dy = sy[i] - qy;
                const float dz = sz[i] - qz;
                const float d2 = dx * dx + dy * dy + dz * dz;
                if (d2 < r2 && cnt < NS) {
                    uint4 q8[8];
                    build_row(feat, pbase + t0 + i, dx, dy, dz, q8);
                    uint4* dst = grow + (size_t)cnt * 8;
                    #pragma unroll
                    for (int u = 0; u < 8; ++u) dst[u] = q8[u];
                    if (cnt == 0) {
                        #pragma unroll
                        for (int u = 0; u < 8; ++u) row0q[u] = q8[u];
                    }
                    ++cnt;
                }
            }
        }
        // barrier (protects LDS reuse) + whole-block early exit
        if (__syncthreads_count(cnt >= NS) == (int)blockDim.x) break;
    }

    if (cnt == 0) {
        const uint4 z = make_uint4(0u, 0u, 0u, 0u);
        for (int s = 0; s < NS; ++s) {
            uint4* dst = grow + (size_t)s * 8;
            #pragma unroll
            for (int u = 0; u < 8; ++u) dst[u] = z;
        }
        return;
    }
    for (int s = cnt; s < NS; ++s) {       // unfilled slots repeat first found
        uint4* dst = grow + (size_t)s * 8;
        #pragma unroll
        for (int u = 0; u < 8; ++u) dst[u] = row0q[u];
    }
}

// ---------------------------------------------------------------------------
// Pad/convert weight (64 x K, f32) -> (64 x 64, bf16), N-major.
// ---------------------------------------------------------------------------
__global__ void prep_w_kernel(const float* __restrict__ w, int K,
                              __bf16* __restrict__ Wp)
{
    const int i = blockIdx.x * 256 + threadIdx.x;
    if (i >= 64 * 64) return;
    const int n = i >> 6, k = i & 63;
    Wp[i] = (k < K) ? (__bf16)w[n * K + k] : (__bf16)0.0f;
}

// ---------------------------------------------------------------------------
// C[M x 64] = A[M x 64 bf16] @ W^T  (W stored N-major 64x64 bf16).
// 256 threads = 8 waves; each wave owns a 16-row tile, 4 N-tiles x 2 K-steps
// of v_wmma_f32_16x16x32_bf16 with f32 accumulators.
// Fragment layouts per CDNA5 ISA 7.12.2.
// ---------------------------------------------------------------------------
__global__ void gemm_bf16_64(const __bf16* __restrict__ A,
                             const __bf16* __restrict__ W,
                             float* __restrict__ C,
                             int Mrows)
{
    const int wave = threadIdx.x >> 5;
    const int lane = threadIdx.x & 31;
    const int half = lane >> 4;       // lane group (0: lanes 0-15, 1: 16-31)
    const int r    = lane & 15;
    const int row0 = (blockIdx.x * 8 + wave) * 16;
    (void)Mrows;

    v8f acc[4] = { v8f{}, v8f{}, v8f{}, v8f{} };

    const __bf16* arow = A + (size_t)(row0 + r) * 64;   // A row for this lane

    #pragma unroll
    for (int ks = 0; ks < 2; ++ks) {
        // A 16x32 bf16 fragment: lane holds K = half*8 + e + 8*(e>=8)  (+32*ks)
        FragBF a;
        a.q[0] = *(const uint4*)(arow + ks * 32 + half * 8);
        a.q[1] = *(const uint4*)(arow + ks * 32 + 16 + half * 8);
        #pragma unroll
        for (int n = 0; n < 4; ++n) {
            // B 32x16 fragment: lane -> column N = n*16 + r, K = half*8 + e + 8*(e>=8) (+32*ks)
            FragBF b;
            const __bf16* wrow = W + (size_t)(n * 16 + r) * 64 + ks * 32 + half * 8;
            b.q[0] = *(const uint4*)(wrow);
            b.q[1] = *(const uint4*)(wrow + 16);
            acc[n] = __builtin_amdgcn_wmma_f32_16x16x32_bf16(
                         false, a.v, false, b.v, (short)0, acc[n], false, false);
        }
    }

    // D 16x16 f32: element v -> (M = half*8 + v, N = n*16 + r)
    #pragma unroll
    for (int n = 0; n < 4; ++n)
        #pragma unroll
        for (int v = 0; v < 8; ++v)
            C[(size_t)(row0 + half * 8 + v) * 64 + n * 16 + r] = acc[n][v];
}

// ---------------------------------------------------------------------------
// Per-channel mean/var, 2-stage deterministic reduction with coalesced reads.
// Stage 1: grid-stride (stride % 64 == 0 keeps each thread's channel fixed),
//          LDS-reduce 256 threads -> 64 per-channel partials per block.
// ---------------------------------------------------------------------------
__global__ void col_stats_stage1(const float* __restrict__ H, int total,
                                 float* __restrict__ psum, float* __restrict__ psq)
{
    __shared__ float ss[256], sq[256];
    const int tid  = threadIdx.x;
    const int gtid = blockIdx.x * 256 + tid;
    const int stride = STAT_BLOCKS * 256;      // multiple of 64
    float s = 0.f, q2 = 0.f;
    for (int i = gtid; i < total; i += stride) {
        const float v = H[i];
        s += v; q2 += v * v;
    }
    ss[tid] = s; sq[tid] = q2;
    __syncthreads();
    if (tid < 64) {
        psum[blockIdx.x * 64 + tid] = ss[tid] + ss[tid + 64] + ss[tid + 128] + ss[tid + 192];
        psq [blockIdx.x * 64 + tid] = sq[tid] + sq[tid + 64] + sq[tid + 128] + sq[tid + 192];
    }
}

// Stage 2: one block per channel, fixed-tree reduction of 256 partials.
__global__ void col_stats_stage2(const float* __restrict__ psum,
                                 const float* __restrict__ psq,
                                 int Mrows,
                                 float* __restrict__ mean, float* __restrict__ var)
{
    __shared__ float ss[256], sq[256];
    const int c = blockIdx.x;
    const int t = threadIdx.x;
    ss[t] = psum[t * 64 + c];
    sq[t] = psq [t * 64 + c];
    __syncthreads();
    for (int o = 128; o > 0; o >>= 1) {
        if (t < o) { ss[t] += ss[t + o]; sq[t] += sq[t + o]; }
        __syncthreads();
    }
    if (t == 0) {
        const float inv = 1.0f / (float)Mrows;
        const float mu  = ss[0] * inv;
        mean[c] = mu;
        var[c]  = fmaxf(sq[0] * inv - mu * mu, 0.f);
    }
}

// ---------------------------------------------------------------------------
// O = bf16(relu(BN(H))): 8 channels per thread, one b128 store.
// ---------------------------------------------------------------------------
__global__ void bn_relu_bf16_kernel(const float* __restrict__ H,
                                    const float* __restrict__ mean,
                                    const float* __restrict__ var,
                                    const float* __restrict__ gamma,
                                    const float* __restrict__ beta,
                                    __bf16* __restrict__ O, int nrows)
{
    const int t   = blockIdx.x * 256 + threadIdx.x;
    const int row = t >> 3;
    if (row >= nrows) return;
    const int c0 = (t & 7) * 8;
    const float* h = H + (size_t)row * 64 + c0;
    unsigned w[4];
    #pragma unroll
    for (int j = 0; j < 4; ++j) {
        const int ca = c0 + 2 * j, cb = ca + 1;
        const float a = fmaxf((h[2*j]   - mean[ca]) * rsqrtf(var[ca] + EPSV) * gamma[ca] + beta[ca], 0.f);
        const float b = fmaxf((h[2*j+1] - mean[cb]) * rsqrtf(var[cb] + EPSV) * gamma[cb] + beta[cb], 0.f);
        w[j] = pk2(a, b);
    }
    *(uint4*)(O + (size_t)row * 64 + c0) = make_uint4(w[0], w[1], w[2], w[3]);
}

// ---------------------------------------------------------------------------
// BN + ReLU + max over ns samples; scatter into concat layout
// out[(roi * TOTCOLS + colbase + grid_slot) * 64 + c].
// ---------------------------------------------------------------------------
__global__ void bn_relu_maxpool_out_kernel(const float* __restrict__ H,
                                           const float* __restrict__ mean,
                                           const float* __restrict__ var,
                                           const float* __restrict__ gamma,
                                           const float* __restrict__ beta,
                                           float* __restrict__ out,
                                           int M, int ns, int gridcube, int colbase)
{
    const int i = blockIdx.x * 256 + threadIdx.x;
    if (i >= M * 64) return;
    const int m = i >> 6, c = i & 63;
    const float mu = mean[c];
    const float rs = rsqrtf(var[c] + EPSV);
    const float ga = gamma[c], be = beta[c];
    const float* hrow = H + (size_t)m * ns * 64;
    float best = 0.f;                    // relu outputs are >= 0
    for (int s = 0; s < ns; ++s) {
        const float v = fmaxf((hrow[(size_t)s * 64 + c] - mu) * rs * ga + be, 0.f);
        best = fmaxf(best, v);
    }
    const int roi = m / gridcube, g = m % gridcube;
    out[((size_t)roi * TOTCOLS + colbase + g) * 64 + c] = best;
}

// ---------------------------------------------------------------------------
extern "C" void kernel_launch(void* const* d_in, const int* in_sizes, int n_in,
                              void* d_out, int out_size, void* d_ws, size_t ws_size,
                              hipStream_t stream)
{
    (void)in_sizes; (void)n_in; (void)out_size; (void)ws_size;

    const float* xyz  = (const float*)d_in[0];
    const float* feat = (const float*)d_in[1];
    const float* nx1  = (const float*)d_in[2];
    const float* nr1  = (const float*)d_in[3];
    const float* nx2  = (const float*)d_in[4];
    const float* nr2  = (const float*)d_in[5];
    const float* w1a  = (const float*)d_in[6];
    const float* g1a  = (const float*)d_in[7];
    const float* b1a  = (const float*)d_in[8];
    const float* w1b  = (const float*)d_in[9];
    const float* g1b  = (const float*)d_in[10];
    const float* b1b  = (const float*)d_in[11];
    const float* w2a  = (const float*)d_in[12];
    const float* g2a  = (const float*)d_in[13];
    const float* b2a  = (const float*)d_in[14];
    const float* w2b  = (const float*)d_in[15];
    const float* g2b  = (const float*)d_in[16];
    const float* b2b  = (const float*)d_in[17];
    float* out = (float*)d_out;

    // Workspace layout (sized for level 2, reused by level 1):
    //   G : 524288 x 64 bf16 (67.1 MB), H : 524288 x 64 f32 (134.2 MB),
    //   Wa/Wb 64x64 bf16, mean/var 64 f32, psum/psq 256x64 f32.
    char* ws = (char*)d_ws;
    const size_t G_CAP = (size_t)524288 * 64 * sizeof(__bf16);
    const size_t H_CAP = (size_t)524288 * 64 * sizeof(float);
    __bf16* Gbuf = (__bf16*)ws;
    float*  Hbuf = (float*)(ws + G_CAP);
    __bf16* Wa   = (__bf16*)(ws + G_CAP + H_CAP);
    __bf16* Wb   = Wa + 64 * 64;
    float*  mean = (float*)(Wb + 64 * 64);
    float*  var  = mean + 64;
    float*  psum = var + 64;
    float*  psq  = psum + STAT_BLOCKS * 64;

    // ---------------- Level 1: ns=16, grid 3^3=27 ----------------
    {
        const int Mq = 128 * 27;        // 3456 queries per batch
        const int M  = BATCH * Mq;      // 6912
        const int rows = M * 16;        // 110592 (multiple of 128)
        prep_w_kernel<<<16, 256, 0, stream>>>(w1a, 35, Wa);
        prep_w_kernel<<<16, 256, 0, stream>>>(w1b, 64, Wb);
        ball_gather_kernel<16><<<M / 128, 128, 0, stream>>>(xyz, feat, nx1, nr1, Gbuf, Mq);
        gemm_bf16_64<<<rows / 128, 256, 0, stream>>>(Gbuf, Wa, Hbuf, rows);
        col_stats_stage1<<<STAT_BLOCKS, 256, 0, stream>>>(Hbuf, rows * 64, psum, psq);
        col_stats_stage2<<<64, 256, 0, stream>>>(psum, psq, rows, mean, var);
        bn_relu_bf16_kernel<<<(rows * 8) / 256, 256, 0, stream>>>(Hbuf, mean, var, g1a, b1a, Gbuf, rows);
        gemm_bf16_64<<<rows / 128, 256, 0, stream>>>(Gbuf, Wb, Hbuf, rows);
        col_stats_stage1<<<STAT_BLOCKS, 256, 0, stream>>>(Hbuf, rows * 64, psum, psq);
        col_stats_stage2<<<64, 256, 0, stream>>>(psum, psq, rows, mean, var);
        bn_relu_maxpool_out_kernel<<<(M * 64) / 256, 256, 0, stream>>>(
            Hbuf, mean, var, g1b, b1b, out, M, 16, 27, 0);
    }

    // ---------------- Level 2: ns=32, grid 4^3=64 ----------------
    {
        const int Mq = 128 * 64;        // 8192 queries per batch
        const int M  = BATCH * Mq;      // 16384
        const int rows = M * 32;        // 524288 (multiple of 128)
        prep_w_kernel<<<16, 256, 0, stream>>>(w2a, 35, Wa);
        prep_w_kernel<<<16, 256, 0, stream>>>(w2b, 64, Wb);
        ball_gather_kernel<32><<<M / 128, 128, 0, stream>>>(xyz, feat, nx2, nr2, Gbuf, Mq);
        gemm_bf16_64<<<rows / 128, 256, 0, stream>>>(Gbuf, Wa, Hbuf, rows);
        col_stats_stage1<<<STAT_BLOCKS, 256, 0, stream>>>(Hbuf, rows * 64, psum, psq);
        col_stats_stage2<<<64, 256, 0, stream>>>(psum, psq, rows, mean, var);
        bn_relu_bf16_kernel<<<(rows * 8) / 256, 256, 0, stream>>>(Hbuf, mean, var, g2a, b2a, Gbuf, rows);
        gemm_bf16_64<<<rows / 128, 256, 0, stream>>>(Gbuf, Wb, Hbuf, rows);
        col_stats_stage1<<<STAT_BLOCKS, 256, 0, stream>>>(Hbuf, rows * 64, psum, psq);
        col_stats_stage2<<<64, 256, 0, stream>>>(psum, psq, rows, mean, var);
        bn_relu_maxpool_out_kernel<<<(M * 64) / 256, 256, 0, stream>>>(
            Hbuf, mean, var, g2b, b2b, out, M, 32, 64, 27);
    }
}
